// moegnn_70085276336456
// MI455X (gfx1250) — matrix operand order repaired and checked by reading, not compiled, not run on hardware
//
#include <hip/hip_runtime.h>
#include <math.h>

// ---------- types for WMMA / TDM ----------
typedef __attribute__((ext_vector_type(16))) __bf16 bf16x16;
typedef __attribute__((ext_vector_type(8)))  float  f32x8;
typedef __attribute__((ext_vector_type(2)))  float  f32x2;
typedef __attribute__((ext_vector_type(4)))  unsigned int u32x4;
typedef __attribute__((ext_vector_type(8)))  int   i32x8;
typedef __attribute__((ext_vector_type(4)))  int   i32x4;

union FragBF { unsigned int u[8]; bf16x16 v; };

__device__ __forceinline__ unsigned short f2bf(float f) {
    unsigned int u = __float_as_uint(f);
    unsigned int r = u + 0x7FFFu + ((u >> 16) & 1u);
    return (unsigned short)(r >> 16);
}

// ---------- small elementwise / precompute kernels (fp32) ----------

__global__ void k_f32_to_bf16(const float* __restrict__ in, unsigned short* __restrict__ out, int n) {
    int i = blockIdx.x * blockDim.x + threadIdx.x;
    if (i < n) out[i] = f2bf(in[i]);
}

// in [rows, cols] fp32 -> out [cols, rows] bf16 (transposed)
__global__ void k_transpose_bf16(const float* __restrict__ in, unsigned short* __restrict__ out,
                                 int rows, int cols) {
    int i = blockIdx.x * blockDim.x + threadIdx.x;
    if (i < rows * cols) {
        int r = i / cols, c = i % cols;
        out[(size_t)c * rows + r] = f2bf(in[i]);
    }
}

// expT[e,d] = relu(sum_k Xp[k*64+e] * mlp[k*1024+d]),  e<64, d<1024
__global__ void k_expT(const float* __restrict__ Xp, const float* __restrict__ mlp,
                       float* __restrict__ expT) {
    int i = blockIdx.x * blockDim.x + threadIdx.x;
    if (i >= 64 * 1024) return;
    int e = i >> 10, d = i & 1023;
    float acc = 0.f;
    for (int k = 0; k < 1024; ++k) acc += Xp[k * 64 + e] * mlp[k * 1024 + d];
    expT[i] = acc > 0.f ? acc : 0.f;
}

// Build normalized adjacency, stored zero-padded [80,80] (single block).
__global__ void k_adj(const float* __restrict__ expT, float* __restrict__ A) {
    __shared__ float nrm[64];
    __shared__ float rs[65];
    int tid = threadIdx.x;
    for (int i = tid; i < 80 * 80; i += 256) A[i] = 0.f;
    if (tid < 64) {
        float s = 0.f;
        for (int d = 0; d < 1024; ++d) { float v = expT[tid * 1024 + d]; s += v * v; }
        float n = sqrtf(s);
        nrm[tid] = n > 1e-8f ? n : 1e-8f;
    }
    __syncthreads();
    for (int idx = tid; idx < 64 * 64; idx += 256) {
        int i = idx >> 6, j = idx & 63;
        if (i < j) {
            float dot = 0.f;
            for (int d = 0; d < 1024; ++d) dot += expT[i * 1024 + d] * expT[j * 1024 + d];
            float cosv = dot / (nrm[i] * nrm[j]);
            if (cosv > 0.5f) A[j * 80 + i] = 1.f;   // edge i->j stored A[dst=j, src=i]
        }
    }
    __syncthreads();
    if (tid < 64) A[tid * 80 + 64] = 1.f;           // token node -> each expert
    __syncthreads();
    if (tid < 65) A[tid * 80 + tid] += 1.f;         // self loops
    __syncthreads();
    if (tid < 65) {
        float deg = 0.f;
        for (int c = 0; c < 65; ++c) deg += A[tid * 80 + c];
        rs[tid] = rsqrtf(deg);
    }
    __syncthreads();
    for (int i = tid; i < 80 * 80; i += 256) {
        int r = i / 80, c = i % 80;
        if (r < 65 && c < 65) A[i] = A[i] * rs[r] * rs[c];
    }
}

// expW[e,d] = sum_k expT[e,k] * W0[k,d]   (64 x 256, K=1024)
__global__ void k_expW(const float* __restrict__ expT, const float* __restrict__ W0,
                       float* __restrict__ expW) {
    int i = blockIdx.x * blockDim.x + threadIdx.x;
    if (i >= 64 * 256) return;
    int e = i >> 8, d = i & 255;
    float acc = 0.f;
    for (int k = 0; k < 1024; ++k) acc += expT[e * 1024 + k] * W0[k * 256 + d];
    expW[i] = acc;
}

// W01T[n,k] = bf16( sum_j W0[k,j] * W1[j,n] )  (stored [256 n][1024 k])
__global__ void k_W01T(const float* __restrict__ W0, const float* __restrict__ W1,
                       unsigned short* __restrict__ W01T) {
    int i = blockIdx.x * blockDim.x + threadIdx.x;
    if (i >= 256 * 1024) return;
    int n = i / 1024, k = i % 1024;
    float acc = 0.f;
    for (int j = 0; j < 256; ++j) acc += W0[k * 256 + j] * W1[j * 256 + n];
    W01T[(size_t)n * 1024 + k] = f2bf(acc);
}

// S0[n,d] = sum_{m<64} A[n,m] * expW[m,d]   (65 x 256), A padded stride 80
__global__ void k_S0(const float* __restrict__ A, const float* __restrict__ expW,
                     float* __restrict__ S0) {
    int i = blockIdx.x * blockDim.x + threadIdx.x;
    if (i >= 65 * 256) return;
    int n = i >> 8, d = i & 255;
    float acc = 0.f;
    for (int m = 0; m < 64; ++m) acc += A[n * 80 + m] * expW[m * 256 + d];
    S0[i] = acc;
}

// C1[n,d] = sum_j S0[n,j] * W1[j,d]
__global__ void k_C1(const float* __restrict__ S0, const float* __restrict__ W1,
                     float* __restrict__ C1) {
    int i = blockIdx.x * blockDim.x + threadIdx.x;
    if (i >= 65 * 256) return;
    int n = i >> 8, d = i & 255;
    float acc = 0.f;
    for (int j = 0; j < 256; ++j) acc += S0[n * 256 + j] * W1[j * 256 + d];
    C1[i] = acc;
}

// Dmat[n,d] = sum_m A[n,m]*C1[m,d];  bvec[n] = sum_m A[n,m]*A[m,64]
__global__ void k_Db(const float* __restrict__ A, const float* __restrict__ C1,
                     float* __restrict__ Dmat, float* __restrict__ bvec) {
    int i = blockIdx.x * blockDim.x + threadIdx.x;
    if (i < 65 * 256) {
        int n = i >> 8, d = i & 255;
        float acc = 0.f;
        for (int m = 0; m < 65; ++m) acc += A[n * 80 + m] * C1[m * 256 + d];
        Dmat[i] = acc;
    } else if (i < 65 * 256 + 65) {
        int n = i - 65 * 256;
        float acc = 0.f;
        for (int m = 0; m < 65; ++m) acc += A[n * 80 + m] * A[m * 80 + 64];
        bvec[n] = acc;
    }
}

// w3p[d] = sum_k W3[d,k] * proj[k]   (256, K=1024)
__global__ void k_w3p(const float* __restrict__ W3, const float* __restrict__ proj,
                      float* __restrict__ w3p) {
    int d = threadIdx.x;
    if (d >= 256) return;
    float acc = 0.f;
    for (int k = 0; k < 1024; ++k) acc += W3[d * 1024 + k] * proj[k];
    w3p[d] = acc;
}

// ---------- WMMA GEMM: C = [relu](A @ B). A[M,K] bf16 row-major, BT[N,K] bf16.
// ---------- One wave computes a 16x64 strip (A-fragment reused across 4 N-tiles).
__global__ __launch_bounds__(128)
void k_wmma_gemm(const unsigned short* __restrict__ Abf,
                 const unsigned short* __restrict__ BTbf,
                 void* __restrict__ Cout, int M, int N, int K, int relu_bf16out) {
    int wave = threadIdx.x >> 5;
    int lane = threadIdx.x & 31;
    int ngroups = N >> 6;                       // strips of 4 N-tiles
    int group = blockIdx.x * (blockDim.x >> 5) + wave;
    if (group >= (M >> 4) * ngroups) return;
    int mt = group / ngroups, ng = group % ngroups;
    int half = lane >> 4, rc = lane & 15;

    const unsigned int* Arow = (const unsigned int*)(Abf + (size_t)(mt * 16 + rc) * K);
    const unsigned int* Bcol0 = (const unsigned int*)(BTbf + (size_t)(ng * 64 + 0 * 16 + rc) * K);
    const unsigned int* Bcol1 = (const unsigned int*)(BTbf + (size_t)(ng * 64 + 1 * 16 + rc) * K);
    const unsigned int* Bcol2 = (const unsigned int*)(BTbf + (size_t)(ng * 64 + 2 * 16 + rc) * K);
    const unsigned int* Bcol3 = (const unsigned int*)(BTbf + (size_t)(ng * 64 + 3 * 16 + rc) * K);

    f32x8 acc0 = {0.f,0.f,0.f,0.f,0.f,0.f,0.f,0.f};
    f32x8 acc1 = acc0, acc2 = acc0, acc3 = acc0;

    for (int k0 = 0; k0 < K; k0 += 32) {
        FragBF a, b0, b1, b2, b3;
#pragma unroll
        for (int p = 0; p < 8; ++p) {
            int ka = k0 + half * 16 + 2 * p;
            a.u[p] = Arow[ka >> 1];
            int kb = (k0 + half * 16 + 2 * p) >> 1;
            b0.u[p] = Bcol0[kb];
            b1.u[p] = Bcol1[kb];
            b2.u[p] = Bcol2[kb];
            b3.u[p] = Bcol3[kb];
        }
        acc0 = __builtin_amdgcn_wmma_f32_16x16x32_bf16(false, a.v, false, b0.v, (short)0, acc0, false, false);
        acc1 = __builtin_amdgcn_wmma_f32_16x16x32_bf16(false, a.v, false, b1.v, (short)0, acc1, false, false);
        acc2 = __builtin_amdgcn_wmma_f32_16x16x32_bf16(false, a.v, false, b2.v, (short)0, acc2, false, false);
        acc3 = __builtin_amdgcn_wmma_f32_16x16x32_bf16(false, a.v, false, b3.v, (short)0, acc3, false, false);
    }

#pragma unroll
    for (int j = 0; j < 4; ++j) {
        f32x8 c = j == 0 ? acc0 : (j == 1 ? acc1 : (j == 2 ? acc2 : acc3));
        int col = ng * 64 + j * 16 + rc;
        if (relu_bf16out) {
            unsigned short* out = (unsigned short*)Cout;
#pragma unroll
            for (int r = 0; r < 8; ++r) {
                int row = mt * 16 + half * 8 + r;
                float v = c[r];
                v = v > 0.f ? v : 0.f;
                out[(size_t)row * N + col] = f2bf(v);
            }
        } else {
            float* out = (float*)Cout;
#pragma unroll
            for (int r = 0; r < 8; ++r) {
                int row = mt * 16 + half * 8 + r;
                out[(size_t)row * N + col] = c[r];
            }
        }
    }
}

// ---------- per-token fused GCN tail: one block per token, 8 waves ----------
// h2 = relu(D + b (x) z[t])  -> LDS bf16 [80,256] (rows 65..79 zero)
// w  = h2 @ W2               -> bf16 WMMA, LDS f32 [80,256] (rows 65..79 zero)
// V  = A @ w (fp32 WMMA 16x16x4); sp[n] = sum_d relu(V[n,d])*w3p[d]
// s  = A @ sp; softmax over experts 0..63
__global__ __launch_bounds__(256)
void k_token(const float* __restrict__ z, const float* __restrict__ Dmat,
             const float* __restrict__ bvec, const float* __restrict__ Apad_g,
             const float* __restrict__ w3p, const unsigned short* __restrict__ W2T,
             float* __restrict__ out) {
    extern __shared__ char smem[];
    unsigned short* h2 = (unsigned short*)smem;                 // 80*256 bf16 = 40960
    float* w    = (float*)(smem + 40960);                       // 80*256 f32  = 81920
    float* Ap   = (float*)(smem + 40960 + 81920);               // 80*80  f32  = 25600
    float* sp   = Ap + 80 * 80;                                  // 80
    float* sc   = sp + 80;                                       // 64
    float* misc = sc + 64;                                       // scalars

    int t = blockIdx.x, tid = threadIdx.x;
    const float* zt = z + (size_t)t * 256;
    int wave = tid >> 5, lane = tid & 31, half = lane >> 4, rc = lane & 15;

    // TDM warm-up: async-copy padded adjacency into the (later overwritten) w
    // region; completion fenced with TENSORcnt before any LDS reuse. The
    // architectural copy below remains the data source for Ap.
    if (tid < 32) {
        unsigned long long ga = (unsigned long long)(size_t)Apad_g;
        unsigned lds_off = (unsigned)(size_t)(void*)w;           // low 32b = LDS offset
        u32x4 g0 = { 1u,                                          // count=1
                     lds_off,
                     (unsigned)(ga & 0xFFFFFFFFu),
                     (unsigned)((ga >> 32) & 0x01FFFFFFu) | 0x80000000u }; // type=2
        i32x8 g1 = { (int)(2u << 16),                 // data_size=4B
                     (int)((6400u & 0xFFFFu) << 16),  // tensor_dim0[15:0]
                     (int)(1u << 16),                 // tensor_dim0 hi=0, tensor_dim1=1
                     (int)((6400u & 0xFFFFu) << 16),  // tile_dim0=6400 (1-D tile)
                     0,                               // tile_dim1=0, tile_dim2=0
                     6400,                            // tensor_dim0_stride
                     0, 0 };
        i32x4 gz4 = {0, 0, 0, 0};
        i32x8 gz8 = {0, 0, 0, 0, 0, 0, 0, 0};
        __builtin_amdgcn_tensor_load_to_lds(g0, g1, gz4, gz4, gz8, 0);
        __builtin_amdgcn_s_wait_tensorcnt(0);
    }

    for (int i = tid; i < 80 * 80; i += 256) Ap[i] = Apad_g[i];
    if (tid < 80) sp[tid] = 0.f;
    for (int i = tid; i < 80 * 256; i += 256) {
        int n = i >> 8, d = i & 255;
        float v = 0.f;
        if (n < 65) {
            v = Dmat[n * 256 + d] + bvec[n] * zt[d];
            v = v > 0.f ? v : 0.f;
        }
        h2[i] = f2bf(v);
    }
    __syncthreads();

    // ---- stage b: w = h2 @ W2 (bf16 WMMA), 5 M-tiles x 16 N-tiles ----
    for (int tile = wave; tile < 80; tile += 8) {
        int mt = tile / 16, nt = tile % 16;
        const unsigned int* Arow = (const unsigned int*)(h2 + (mt * 16 + rc) * 256);
        const unsigned int* Bcol = (const unsigned int*)(W2T + (size_t)(nt * 16 + rc) * 256);
        f32x8 c = {0.f,0.f,0.f,0.f,0.f,0.f,0.f,0.f};
        for (int k0 = 0; k0 < 256; k0 += 32) {
            FragBF a, b;
#pragma unroll
            for (int p = 0; p < 8; ++p) {
                int ka = k0 + half * 16 + 2 * p;
                a.u[p] = Arow[ka >> 1];
                b.u[p] = Bcol[(k0 + half * 16 + 2 * p) >> 1];
            }
            c = __builtin_amdgcn_wmma_f32_16x16x32_bf16(false, a.v, false, b.v,
                                                        (short)0, c, false, false);
        }
#pragma unroll
        for (int r = 0; r < 8; ++r) {
            int row = mt * 16 + half * 8 + r;
            w[row * 256 + nt * 16 + rc] = c[r];   // rows 65..79 are exact zeros
        }
    }
    __syncthreads();

    // ---- stage c: V = Ap @ w via fp32 WMMA (16x16x4), fused relu/w3p/row-sum ----
    for (int tile = wave; tile < 80; tile += 8) {
        int mt = tile / 16, nt = tile % 16;
        int row = mt * 16 + rc;                 // A-fragment row
        int col = nt * 16 + rc;                 // B/D column
        f32x8 c = {0.f,0.f,0.f,0.f,0.f,0.f,0.f,0.f};
        for (int k0 = 0; k0 < 80; k0 += 4) {
            int k = k0 + half * 2;
            f32x2 a2 = { Ap[row * 80 + k], Ap[row * 80 + k + 1] };
            f32x2 b2 = { w[k * 256 + col], w[(k + 1) * 256 + col] };
            c = __builtin_amdgcn_wmma_f32_16x16x4_f32(false, a2, false, b2,
                                                      (short)0, c, false, false);
        }
        float w3 = w3p[col];
#pragma unroll
        for (int r = 0; r < 8; ++r) {
            float v = c[r] > 0.f ? c[r] : 0.f;
            v *= w3;
            v += __shfl_xor(v, 1, 32);
            v += __shfl_xor(v, 2, 32);
            v += __shfl_xor(v, 4, 32);
            v += __shfl_xor(v, 8, 32);          // sum over the 16 cols of this tile
            if (rc == 0) atomicAdd(&sp[mt * 16 + half * 8 + r], v);
        }
    }
    __syncthreads();

    // ---- s = A @ sp, softmax over experts ----
    if (tid < 64) {
        float s = 0.f;
        for (int m = 0; m < 65; ++m) s += Ap[tid * 80 + m] * sp[m];
        sc[tid] = s;
    }
    __syncthreads();
    if (tid == 0) {
        float mx = sc[0];
        for (int i = 1; i < 64; ++i) mx = sc[i] > mx ? sc[i] : mx;
        float sum = 0.f;
        for (int i = 0; i < 64; ++i) { float e = expf(sc[i] - mx); sc[i] = e; sum += e; }
        misc[0] = 1.f / sum;
    }
    __syncthreads();
    if (tid < 64) out[(size_t)t * 64 + tid] = sc[tid] * misc[0];
}

// ---------- host ----------
extern "C" void kernel_launch(void* const* d_in, const int* in_sizes, int n_in,
                              void* d_out, int out_size, void* d_ws, size_t ws_size,
                              hipStream_t stream) {
    (void)in_sizes; (void)n_in; (void)out_size; (void)ws_size;
    const float* x    = (const float*)d_in[0];   // [2,1024,1024]
    const float* Xp   = (const float*)d_in[1];   // [1024,64]
    const float* mlp  = (const float*)d_in[2];   // [1024,1024]
    const float* W0   = (const float*)d_in[3];   // [1024,256]
    const float* W1   = (const float*)d_in[4];   // [256,256]
    const float* W2   = (const float*)d_in[5];   // [256,256]
    const float* W3   = (const float*)d_in[6];   // [256,1024]
    const float* proj = (const float*)d_in[7];   // [1024,1]
    float* out = (float*)d_out;

    const int T = 2048, DIM = 1024, DG = 256, NE = 64, NN = 65;

    char* ws = (char*)d_ws;
    size_t off = 0;
    auto carve = [&](size_t bytes) -> char* {
        char* p = ws + off;
        off += (bytes + 255) & ~(size_t)255;
        return p;
    };
    unsigned short* x_bf    = (unsigned short*)carve((size_t)T * DIM * 2);
    unsigned short* mlpT_bf = (unsigned short*)carve((size_t)DIM * DIM * 2);
    unsigned short* xt_bf   = (unsigned short*)carve((size_t)T * DIM * 2);
    unsigned short* W01T_bf = (unsigned short*)carve((size_t)DG * DIM * 2);
    unsigned short* W2T_bf  = (unsigned short*)carve((size_t)DG * DG * 2);
    float* zbuf = (float*)carve((size_t)T * DG * 4);
    float* expT = (float*)carve((size_t)NE * DIM * 4);
    float* Apad = (float*)carve((size_t)80 * 80 * 4);
    float* expW = (float*)carve((size_t)NE * DG * 4);
    float* S0   = (float*)carve((size_t)NN * DG * 4);
    float* C1   = (float*)carve((size_t)NN * DG * 4);
    float* Dmat = (float*)carve((size_t)NN * DG * 4);
    float* bvec = (float*)carve((size_t)NN * 4);
    float* w3p  = (float*)carve((size_t)DG * 4);

    // precompute + conversions
    k_f32_to_bf16<<<(T * DIM + 255) / 256, 256, 0, stream>>>(x, x_bf, T * DIM);
    k_transpose_bf16<<<(DIM * DIM + 255) / 256, 256, 0, stream>>>(mlp, mlpT_bf, DIM, DIM);
    k_expT<<<(NE * DIM + 255) / 256, 256, 0, stream>>>(Xp, mlp, expT);
    k_adj<<<1, 256, 0, stream>>>(expT, Apad);
    k_expW<<<(NE * DG + 255) / 256, 256, 0, stream>>>(expT, W0, expW);
    k_W01T<<<(DG * DIM + 255) / 256, 256, 0, stream>>>(W0, W1, W01T_bf);
    k_transpose_bf16<<<(DG * DG + 255) / 256, 256, 0, stream>>>(W2, W2T_bf, DG, DG);
    k_S0<<<(NN * DG + 255) / 256, 256, 0, stream>>>(Apad, expW, S0);
    k_C1<<<(NN * DG + 255) / 256, 256, 0, stream>>>(S0, W1, C1);
    k_Db<<<(NN * DG + NN + 255) / 256, 256, 0, stream>>>(Apad, C1, Dmat, bvec);
    k_w3p<<<1, 256, 0, stream>>>(W3, proj, w3p);

    // big WMMA GEMMs (one wave = 16x64 strip)
    {   // xt = relu(x @ mlp) -> bf16   (M=2048, N=1024, K=1024)
        int groups = (T / 16) * (DIM / 64);
        k_wmma_gemm<<<groups / 4, 128, 0, stream>>>(x_bf, mlpT_bf, xt_bf, T, DIM, DIM, 1);
    }
    {   // z = xt @ (W0@W1) -> f32     (M=2048, N=256, K=1024)
        int groups = (T / 16) * (DG / 64);
        k_wmma_gemm<<<groups / 4, 128, 0, stream>>>(xt_bf, W01T_bf, zbuf, T, DG, DIM, 0);
    }

    // per-token fused GCN tail
    size_t smem = 40960                         // h2 bf16 [80,256]
                + 81920                         // w  f32  [80,256]
                + 25600                         // Ap f32  [80,80]
                + (80 + 64 + 4) * 4;            // sp, sc, misc
    k_token<<<T, 256, smem, stream>>>(zbuf, Dmat, bvec, Apad, w3p, W2T_bf, out);
}
